// TransformerDecoder_22600117911637
// MI455X (gfx1250) — matrix operand order repaired and checked by reading, not compiled, not run on hardware
//
#include <hip/hip_runtime.h>
#include <math.h>

typedef unsigned short ushort_t;
typedef __attribute__((ext_vector_type(16))) __bf16 bf16x16;
typedef __attribute__((ext_vector_type(8)))  float  f32x8;

constexpr int BATCH = 8, SEQ = 512, DIM = 1024, VOCAB = 32000;
constexpr int LAYERS = 6, HEADS = 16, DFF = 4096, HDIM = 64;
constexpr float NEGVAL = -1e8f;

// GEMM block tile
constexpr int GM = 128, GN = 128, GK = 32;

// Pack two floats to two bf16 (round-to-nearest) in one dword:
// one v_add per input + one v_perm_b32 (bytes 7,6 of hi | bytes 3,2 of lo).
__device__ __forceinline__ unsigned packbf2(float lo, float hi) {
    unsigned ul = __float_as_uint(lo) + 0x8000u;
    unsigned uh = __float_as_uint(hi) + 0x8000u;
    return __builtin_amdgcn_perm(uh, ul, 0x07060302u);
}

union BF16Frag { ushort_t u[16]; unsigned w[8]; uint4 q[2]; bf16x16 v; };

__device__ __forceinline__ f32x8 zero8() {
    f32x8 z;
#pragma unroll
    for (int i = 0; i < 8; ++i) z[i] = 0.0f;
    return z;
}

__device__ __forceinline__ f32x8 wmma_bf16(const BF16Frag& a, const BF16Frag& b, f32x8 c) {
    return __builtin_amdgcn_wmma_f32_16x16x32_bf16(
        /*neg_a=*/false, a.v, /*neg_b=*/false, b.v,
        /*c_mod=*/(short)0, c, /*reuse_a=*/false, /*reuse_b=*/false);
}

// A fragment (16x32 bf16). Per ISA: lane half h=L>>4, per-lane elements are the
// two contiguous k-chunks [KOFF+8h, +8) and [KOFF+16+8h, +8)  -> 2x ds_load_b128.
#define LOAD_A_FRAG(dst, ARR, ROW0, KOFF) do {                                  \
    const ushort_t* _p = &ARR[(ROW0) + (lane & 15)][(KOFF) + (lane >> 4) * 8];  \
    (dst).q[0] = *(const uint4*)_p;                                             \
    (dst).q[1] = *(const uint4*)(_p + 16);                                      \
  } while (0)

// B fragment (32x16 bf16) from a transposed tile ARR[n][k]: lane reads the two
// k-chunks [KOFF+8h, +8) and [KOFF+16+8h, +8) of row n  -> 2x ds_load_b128.
#define LOAD_B_FRAG(dst, ARR, COL0, KOFF) do {                                  \
    const ushort_t* _p = &ARR[(COL0) + (lane & 15)][(KOFF) + (lane >> 4) * 8];  \
    (dst).q[0] = *(const uint4*)_p;                                             \
    (dst).q[1] = *(const uint4*)(_p + 16);                                      \
  } while (0)

// ---------------------------------------------------------------------------
// GEMM: C[M,N] = act(A[M,K] @ W[K,N] + bias[N]); fp32 in/out, bf16 WMMA.
// 256 threads (8 waves), tile 128x128x32, double-buffered LDS, one barrier
// per K-step. Wave -> 32x64 patch (8 WMMAs per step).
// ---------------------------------------------------------------------------
__global__ __launch_bounds__(256) void gemm_bias_act(
    const float* __restrict__ A, const float* __restrict__ W,
    const float* __restrict__ bias, float* __restrict__ C,
    int M, int N, int K, int relu_act)
{
    __shared__ ushort_t As[2][GM][GK + 8];   // row-major (k contiguous), 80B stride
    __shared__ ushort_t Bt[2][GN][GK + 8];   // TRANSPOSED: Bt[n][k]

    const int tid  = threadIdx.x;
    const int lane = tid & 31;
    const int wid  = tid >> 5;
    const int m_blk = blockIdx.y * GM;
    const int n_blk = blockIdx.x * GN;
    const int wave_m = (wid >> 1) * 32;   // 4 wave-rows
    const int wave_n = (wid & 1) * 64;    // 2 wave-cols

    f32x8 acc[2][4];
#pragma unroll
    for (int i = 0; i < 2; ++i)
#pragma unroll
        for (int j = 0; j < 4; ++j) acc[i][j] = zero8();

    // A tile 128x32: thread -> row ar, 16 contiguous k at ac (b128 x4 loads)
    const int ar = tid >> 1, ac = (tid & 1) * 16;
    // B tile 32x128: thread -> one column n (lanes cover 128 consecutive n),
    //                16 k-strided rows starting at bkh.
    const int bn_l = tid & 127, bkh = (tid >> 7) * 16;
    const int bn   = n_blk + bn_l;

    float areg[16];
    float breg[16];

    auto stage_load = [&](int kk) {
        const float* asrc = A + (size_t)(m_blk + ar) * K + kk + ac;
#pragma unroll
        for (int q = 0; q < 4; ++q) {
            float4 p = ((const float4*)asrc)[q];
            areg[4*q + 0] = p.x; areg[4*q + 1] = p.y;
            areg[4*q + 2] = p.z; areg[4*q + 3] = p.w;
        }
#pragma unroll
        for (int j = 0; j < 16; ++j)
            breg[j] = W[(size_t)(kk + bkh + j) * N + bn];
    };
    auto stage_store = [&](int buf) {
        union { unsigned w[8]; uint4 q[2]; } ta, tb;
#pragma unroll
        for (int j = 0; j < 8; ++j) ta.w[j] = packbf2(areg[2*j], areg[2*j + 1]);
#pragma unroll
        for (int j = 0; j < 8; ++j) tb.w[j] = packbf2(breg[2*j], breg[2*j + 1]);
        *(uint4*)&As[buf][ar][ac]       = ta.q[0];
        *(uint4*)&As[buf][ar][ac + 8]   = ta.q[1];
        *(uint4*)&Bt[buf][bn_l][bkh]     = tb.q[0];
        *(uint4*)&Bt[buf][bn_l][bkh + 8] = tb.q[1];
    };

    stage_load(0);
    stage_store(0);
    __syncthreads();

    int cur = 0;
    for (int k0 = 0; k0 < K; k0 += GK) {
        const bool has_next = (k0 + GK < K);
        if (has_next) stage_load(k0 + GK);             // overlap with compute
        if (k0 + 2 * GK < K) {                         // global_prefetch_b8
            __builtin_prefetch(A + (size_t)(m_blk + ar) * K + k0 + 2 * GK + ac, 0, 1);
            __builtin_prefetch(W + (size_t)(k0 + 2 * GK + bkh) * N + bn, 0, 1);
        }

        BF16Frag afrag[2];
#pragma unroll
        for (int im = 0; im < 2; ++im)
            LOAD_A_FRAG(afrag[im], As[cur], wave_m + im * 16, 0);
#pragma unroll
        for (int in = 0; in < 4; ++in) {
            BF16Frag bf;
            LOAD_B_FRAG(bf, Bt[cur], wave_n + in * 16, 0);
            acc[0][in] = wmma_bf16(afrag[0], bf, acc[0][in]);
            acc[1][in] = wmma_bf16(afrag[1], bf, acc[1][in]);
        }

        if (has_next) stage_store(cur ^ 1);
        __syncthreads();
        cur ^= 1;
    }

    // Epilogue: D layout -> lane n = L%16, VGPR r -> m = r + 8*(L/16).
#pragma unroll
    for (int im = 0; im < 2; ++im) {
#pragma unroll
        for (int in = 0; in < 4; ++in) {
            int gn = n_blk + wave_n + in * 16 + (lane & 15);
            float bvv = bias[gn];
            int mbase = m_blk + wave_m + im * 16 + (lane >> 4) * 8;
#pragma unroll
            for (int r = 0; r < 8; ++r) {
                float val = acc[im][in][r] + bvv;
                if (relu_act) val = fmaxf(val, 0.0f);
                C[(size_t)(mbase + r) * N + gn] = val;
            }
        }
    }
}

// ---------------------------------------------------------------------------
// Flash attention: block = (b, h, 64-query tile); K/V streamed in 64-key tiles.
// S = Q K^T via WMMA (Ks natural layout serves as B), online softmax in LDS,
// O += P V via WMMA (V staged transposed: Vt[d][key]).
// ---------------------------------------------------------------------------
__global__ __launch_bounds__(256) void flash_attn_kernel(
    const float* __restrict__ Q, const float* __restrict__ Kp,
    const float* __restrict__ Vp, float* __restrict__ O,
    const int* __restrict__ kmask, int causal)
{
    __shared__ ushort_t Qs[64][HDIM + 8];   // Qs[q][d]   (A for scores)
    __shared__ ushort_t Ks[64][HDIM + 8];   // Ks[key][d] (B for scores, n=key,k=d)
    __shared__ ushort_t Vt[HDIM][64 + 8];   // Vt[d][key] (B for PV, n=d,k=key)
    __shared__ float    Sc[64][65];         // fp32 score tile
    __shared__ ushort_t Ps[64][64 + 8];     // bf16 probabilities (A for PV)
    __shared__ float    redm[64][4], reds[64][4];
    __shared__ float    rowm[64], rowl[64], rowc[64];

    const int tid  = threadIdx.x;
    const int lane = tid & 31;
    const int wid  = tid >> 5;
    const int q0 = blockIdx.x * 64;
    const int hh = blockIdx.y;
    const int bb = blockIdx.z;

    const int qsub  = wid >> 1;   // 16-row query strip owned by this wave
    const int kpart = wid & 1;    // 32-wide key half (score phase)
    const int dsub  = wid & 1;    // 32-wide d half (output phase)

    {   // load Q tile (64 x 64) as bf16, row-contiguous
        int r = tid >> 2, cb = (tid & 3) * 16;
        const float* src = Q + (size_t)(bb * SEQ + q0 + r) * DIM + hh * HDIM + cb;
        union { unsigned w[8]; uint4 q[2]; } t;
#pragma unroll
        for (int qq = 0; qq < 4; ++qq) {
            float4 p = ((const float4*)src)[qq];
            t.w[2*qq + 0] = packbf2(p.x, p.y);
            t.w[2*qq + 1] = packbf2(p.z, p.w);
        }
        *(uint4*)&Qs[r][cb]     = t.q[0];
        *(uint4*)&Qs[r][cb + 8] = t.q[1];
    }
    if (tid < 64) { rowm[tid] = -3.0e38f; rowl[tid] = 0.0f; }
    __syncthreads();

    f32x8 o_acc[2];
    o_acc[0] = zero8(); o_acc[1] = zero8();

    const int vd = tid & 63, vkh = (tid >> 6) * 16;   // V transpose staging roles

    for (int kt = 0; kt < SEQ / 64; ++kt) {
        const int kbase = kt * 64;
        {   // K tile: natural row-major
            int r = tid >> 2, cb = (tid & 3) * 16;
            const float* ks = Kp + (size_t)(bb * SEQ + kbase + r) * DIM + hh * HDIM + cb;
            union { unsigned w[8]; uint4 q[2]; } t;
#pragma unroll
            for (int qq = 0; qq < 4; ++qq) {
                float4 p = ((const float4*)ks)[qq];
                t.w[2*qq + 0] = packbf2(p.x, p.y);
                t.w[2*qq + 1] = packbf2(p.z, p.w);
            }
            *(uint4*)&Ks[r][cb]     = t.q[0];
            *(uint4*)&Ks[r][cb + 8] = t.q[1];
        }
        {   // V tile: transposed (thread owns one d column; lanes coalesce over d)
            float vv[16];
#pragma unroll
            for (int j = 0; j < 16; ++j)
                vv[j] = Vp[(size_t)(bb * SEQ + kbase + vkh + j) * DIM + hh * HDIM + vd];
            union { unsigned w[8]; uint4 q[2]; } t;
#pragma unroll
            for (int j = 0; j < 8; ++j) t.w[j] = packbf2(vv[2*j], vv[2*j + 1]);
            *(uint4*)&Vt[vd][vkh]     = t.q[0];
            *(uint4*)&Vt[vd][vkh + 8] = t.q[1];
        }
        __syncthreads();

        // S = Q K^T over this 64-key tile (each wave: 16q x 32k)
        f32x8 s_acc[2];
        s_acc[0] = zero8(); s_acc[1] = zero8();
#pragma unroll
        for (int dstep = 0; dstep < HDIM; dstep += 32) {
            BF16Frag aq, bk0, bk1;
            LOAD_A_FRAG(aq,  Qs, qsub * 16,        dstep);
            LOAD_B_FRAG(bk0, Ks, kpart * 32 + 0,   dstep);
            LOAD_B_FRAG(bk1, Ks, kpart * 32 + 16,  dstep);
            s_acc[0] = wmma_bf16(aq, bk0, s_acc[0]);
            s_acc[1] = wmma_bf16(aq, bk1, s_acc[1]);
        }
        // scale + mask, write fp32 scores to LDS
#pragma unroll
        for (int j = 0; j < 2; ++j) {
            int n  = kpart * 32 + j * 16 + (lane & 15);
            int kg = kbase + n;
            int km = kmask[bb * SEQ + kg];
#pragma unroll
            for (int r = 0; r < 8; ++r) {
                int m  = qsub * 16 + r + (lane >> 4) * 8;
                int qg = q0 + m;
                bool masked = (km != 0) || (causal && (kg > qg));
                Sc[m][n] = masked ? NEGVAL : s_acc[j][r] * 0.125f;  // 1/sqrt(64)
            }
        }
        __syncthreads();

        // online softmax: 4 threads per row, 16 columns each
        const int row = tid >> 2, qtr = tid & 3, c0 = qtr * 16;
        {
            float lmax = -3.0e38f;
#pragma unroll
            for (int c = 0; c < 16; ++c) lmax = fmaxf(lmax, Sc[row][c0 + c]);
            redm[row][qtr] = lmax;
        }
        __syncthreads();
        float nm = fmaxf(fmaxf(redm[row][0], redm[row][1]),
                         fmaxf(redm[row][2], redm[row][3]));
        nm = fmaxf(nm, rowm[row]);
        {
            float lsum = 0.0f;
            unsigned* prow = (unsigned*)&Ps[row][c0];
#pragma unroll
            for (int c = 0; c < 16; c += 2) {
                float p0 = __expf(Sc[row][c0 + c]     - nm);
                float p1 = __expf(Sc[row][c0 + c + 1] - nm);
                lsum += p0 + p1;
                prow[c >> 1] = packbf2(p0, p1);
            }
            reds[row][qtr] = lsum;
        }
        __syncthreads();
        if (qtr == 0) {
            float corr = __expf(rowm[row] - nm);
            rowl[row] = rowl[row] * corr +
                        (reds[row][0] + reds[row][1] + reds[row][2] + reds[row][3]);
            rowm[row] = nm;
            rowc[row] = corr;
        }
        __syncthreads();

        // rescale O accumulator, then O += P V (each wave: 16q x 32d)
#pragma unroll
        for (int j = 0; j < 2; ++j)
#pragma unroll
            for (int r = 0; r < 8; ++r) {
                int m = qsub * 16 + r + (lane >> 4) * 8;
                o_acc[j][r] *= rowc[m];
            }
#pragma unroll
        for (int kstep = 0; kstep < 64; kstep += 32) {
            BF16Frag ap, bv0, bv1;
            LOAD_A_FRAG(ap,  Ps, qsub * 16,       kstep);
            LOAD_B_FRAG(bv0, Vt, dsub * 32 + 0,   kstep);
            LOAD_B_FRAG(bv1, Vt, dsub * 32 + 16,  kstep);
            o_acc[0] = wmma_bf16(ap, bv0, o_acc[0]);
            o_acc[1] = wmma_bf16(ap, bv1, o_acc[1]);
        }
        __syncthreads();
    }

    // normalize and store O
#pragma unroll
    for (int j = 0; j < 2; ++j) {
        int dc = hh * HDIM + dsub * 32 + j * 16 + (lane & 15);
#pragma unroll
        for (int r = 0; r < 8; ++r) {
            int m = qsub * 16 + r + (lane >> 4) * 8;
            float inv = 1.0f / rowl[m];
            O[(size_t)(bb * SEQ + q0 + m) * DIM + dc] = o_acc[j][r] * inv;
        }
    }
}

// ---------------------------------------------------------------------------
// h = LayerNorm(h + a) * g + b ; one block per row (in-place on h).
// ---------------------------------------------------------------------------
__global__ __launch_bounds__(256) void ln_residual_kernel(
    float* __restrict__ h, const float* __restrict__ a,
    const float* __restrict__ g, const float* __restrict__ b)
{
    const int row = blockIdx.x;
    const int tid = threadIdx.x;
    float x[4];
    float s = 0.0f, s2 = 0.0f;
#pragma unroll
    for (int j = 0; j < 4; ++j) {
        int idx = tid + j * 256;
        float v = h[(size_t)row * DIM + idx] + a[(size_t)row * DIM + idx];
        x[j] = v; s += v; s2 += v * v;
    }
    __shared__ float rs[256], rs2[256];
    rs[tid] = s; rs2[tid] = s2;
    __syncthreads();
    for (int st = 128; st > 0; st >>= 1) {
        if (tid < st) { rs[tid] += rs[tid + st]; rs2[tid] += rs2[tid + st]; }
        __syncthreads();
    }
    float mu  = rs[0] * (1.0f / DIM);
    float var = rs2[0] * (1.0f / DIM) - mu * mu;
    float rinv = rsqrtf(var + 1e-5f);
#pragma unroll
    for (int j = 0; j < 4; ++j) {
        int idx = tid + j * 256;
        h[(size_t)row * DIM + idx] = (x[j] - mu) * rinv * g[idx] + b[idx];
    }
}

__global__ __launch_bounds__(256) void embed_kernel(
    const int* __restrict__ x, const float* __restrict__ tok,
    const float* __restrict__ pos, float* __restrict__ h)
{
    const int row = blockIdx.x;           // b*S + s
    const int s = row % SEQ;
    const int t = x[row];
    for (int i = threadIdx.x; i < DIM; i += 256)
        h[(size_t)row * DIM + i] = tok[(size_t)t * DIM + i] * (1.0f / 32.0f)
                                 + pos[(size_t)s * DIM + i];
}

__global__ __launch_bounds__(256) void mask_kernel(
    const int* __restrict__ x, const float* __restrict__ enc,
    int* __restrict__ padm, int* __restrict__ encm)
{
    const int row = blockIdx.x;           // b*S + s
    __shared__ int any_zero;
    if (threadIdx.x == 0) { padm[row] = (x[row] == 0) ? 1 : 0; any_zero = 0; }
    __syncthreads();
    int local = 0;
    for (int i = threadIdx.x; i < DIM; i += 256)
        local |= (enc[(size_t)row * DIM + i] == 0.0f) ? 1 : 0;
    if (local) atomicOr(&any_zero, 1);
    __syncthreads();
    if (threadIdx.x == 0) encm[row] = any_zero;   // 1 => key masked
}

extern "C" void kernel_launch(void* const* d_in, const int* in_sizes, int n_in,
                              void* d_out, int out_size, void* d_ws, size_t ws_size,
                              hipStream_t stream) {
    (void)in_sizes; (void)n_in; (void)out_size; (void)ws_size;

    const int*   x       = (const int*)d_in[0];
    const float* enc_out = (const float*)d_in[1];
    const float* tok     = (const float*)d_in[2];
    const float* pos     = (const float*)d_in[3];
    const float* Wq  = (const float*)d_in[4];  const float* bq  = (const float*)d_in[5];
    const float* Wk  = (const float*)d_in[6];  const float* bk  = (const float*)d_in[7];
    const float* Wv  = (const float*)d_in[8];  const float* bv  = (const float*)d_in[9];
    const float* Wo  = (const float*)d_in[10]; const float* bo  = (const float*)d_in[11];
    const float* ln1g = (const float*)d_in[12]; const float* ln1b = (const float*)d_in[13];
    const float* Wq2 = (const float*)d_in[14]; const float* bq2 = (const float*)d_in[15];
    const float* Wk2 = (const float*)d_in[16]; const float* bk2 = (const float*)d_in[17];
    const float* Wv2 = (const float*)d_in[18]; const float* bv2 = (const float*)d_in[19];
    const float* Wo2 = (const float*)d_in[20]; const float* bo2 = (const float*)d_in[21];
    const float* ln2g = (const float*)d_in[22]; const float* ln2b = (const float*)d_in[23];
    const float* W1  = (const float*)d_in[24]; const float* b1  = (const float*)d_in[25];
    const float* W2  = (const float*)d_in[26]; const float* b2  = (const float*)d_in[27];
    const float* ln3g = (const float*)d_in[28]; const float* ln3b = (const float*)d_in[29];
    const float* Wc  = (const float*)d_in[30]; const float* bc  = (const float*)d_in[31];

    const int M = BATCH * SEQ;                 // 4096
    const size_t ND = (size_t)M * DIM;

    float* h   = (float*)d_ws;
    float* qb  = h  + ND;
    float* kb  = qb + ND;
    float* vb  = kb + ND;
    float* ab  = vb + ND;                      // attention out (pre-projection)
    float* pb  = ab + ND;                      // projection / ffn2 out
    float* ff1 = pb + ND;                      // M x DFF
    int* padm  = (int*)(ff1 + (size_t)M * DFF);
    int* encm  = padm + M;

    embed_kernel<<<M, 256, 0, stream>>>(x, tok, pos, h);
    mask_kernel<<<M, 256, 0, stream>>>(x, enc_out, padm, encm);

    auto gemm = [&](const float* A, const float* Wt, const float* bias, float* C,
                    int m, int n, int k, int act) {
        dim3 grid(n / GN, m / GM);
        gemm_bias_act<<<grid, 256, 0, stream>>>(A, Wt, bias, C, m, n, k, act);
    };

    dim3 agrid(SEQ / 64, HEADS, BATCH);

    for (int l = 0; l < LAYERS; ++l) {
        const size_t wdd = (size_t)l * DIM * DIM;
        const size_t wd  = (size_t)l * DIM;
        // masked self-attention
        gemm(h, Wq + wdd, bq + wd, qb, M, DIM, DIM, 0);
        gemm(h, Wk + wdd, bk + wd, kb, M, DIM, DIM, 0);
        gemm(h, Wv + wdd, bv + wd, vb, M, DIM, DIM, 0);
        flash_attn_kernel<<<agrid, 256, 0, stream>>>(qb, kb, vb, ab, padm, 1);
        gemm(ab, Wo + wdd, bo + wd, pb, M, DIM, DIM, 0);
        ln_residual_kernel<<<M, 256, 0, stream>>>(h, pb, ln1g + wd, ln1b + wd);
        // cross-attention against encoder output
        gemm(h,       Wq2 + wdd, bq2 + wd, qb, M, DIM, DIM, 0);
        gemm(enc_out, Wk2 + wdd, bk2 + wd, kb, M, DIM, DIM, 0);
        gemm(enc_out, Wv2 + wdd, bv2 + wd, vb, M, DIM, DIM, 0);
        flash_attn_kernel<<<agrid, 256, 0, stream>>>(qb, kb, vb, ab, encm, 0);
        gemm(ab, Wo2 + wdd, bo2 + wd, pb, M, DIM, DIM, 0);
        ln_residual_kernel<<<M, 256, 0, stream>>>(h, pb, ln2g + wd, ln2b + wd);
        // FFN
        gemm(h,   W1 + (size_t)l * DIM * DFF, b1 + (size_t)l * DFF, ff1, M, DFF, DIM, 1);
        gemm(ff1, W2 + (size_t)l * DFF * DIM, b2 + wd,              pb,  M, DIM, DFF, 0);
        ln_residual_kernel<<<M, 256, 0, stream>>>(h, pb, ln3g + wd, ln3b + wd);
    }

    // classifier: [4096,1024] @ [1024,32000] + bc -> d_out
    gemm(h, Wc, bc, (float*)d_out, M, VOCAB, DIM, 0);
}
